// OTransformerLayer_47614007443736
// MI455X (gfx1250) — matrix operand order repaired and checked by reading, not compiled, not run on hardware
//
#include <hip/hip_runtime.h>
#include <hip/hip_bf16.h>
#include <math.h>

// Problem sizes (fixed by the reference)
#define BQ   4
#define SQ   1024
#define DQ   1024
#define HQ   16
#define DKQ  64
#define DFQ  4096
#define TOPK 307   // max(1, int(1024*0.3))

#define APITCH 40  // LDS row pitch in bf16 elems (80B = 5*16B, conflict-free)

typedef __attribute__((ext_vector_type(16))) __bf16 v16bf;
typedef __attribute__((ext_vector_type(8)))  float  v8f;
typedef int v4i __attribute__((__vector_size__(16)));
typedef __attribute__((address_space(1))) v4i v4i_g;  // global
typedef __attribute__((address_space(3))) v4i v4i_l;  // LDS

union FragU { uint4 u[2]; v16bf v; };

// ---------------------------------------------------------------------------
// CDNA5 async global->LDS copy (ASYNCcnt) with builtin/asm fallback
// ---------------------------------------------------------------------------
#if defined(__has_builtin)
#  if __has_builtin(__builtin_amdgcn_global_load_async_to_lds_b128)
#    define ASYNC_LDS_BUILTIN 1
#  endif
#  if __has_builtin(__builtin_amdgcn_s_wait_asynccnt)
#    define WAIT_ASYNC_BUILTIN 1
#  endif
#endif

__device__ __forceinline__ void async_load_b128(const void* gp, void* lp) {
#ifdef ASYNC_LDS_BUILTIN
  void* gnc = const_cast<void*>(gp);
  __builtin_amdgcn_global_load_async_to_lds_b128((v4i_g*)gnc, (v4i_l*)lp, 0, 0);
#else
  auto l3 = (__attribute__((address_space(3))) char*)(char*)lp;
  unsigned loff = (unsigned)(size_t)l3;
  asm volatile("global_load_async_to_lds_b128 %0, %1, off"
               :: "v"(loff), "v"(gp) : "memory");
#endif
}

__device__ __forceinline__ void wait_async0() {
#ifdef WAIT_ASYNC_BUILTIN
  __builtin_amdgcn_s_wait_asynccnt(0);
#else
  asm volatile("s_wait_asynccnt 0" ::: "memory");
#endif
}

// ---------------------------------------------------------------------------
// WMMA fragment loaders (wave32, V_WMMA_F32_16X16X32_BF16 layouts per ISA 7.12.2)
// A (16x32, MxK): lanes 0-15 -> M=lane, K = {k0..k0+7} then {k0+16..k0+23}
//                 lanes 16-31 -> M=lane-16, K = {k0+8..} then {k0+24..}
// B (32x16, KxN): lanes 0-15 -> N=lane, K = {k0..k0+7} then {k0+16..k0+23}
//                 lanes 16-31 -> N=lane-16, K = {k0+8..} then {k0+24..}
// ---------------------------------------------------------------------------
__device__ __forceinline__ v16bf lds_frag_a(const __bf16* tile, int mlocal) {
  int lane = threadIdx.x & 31;
  int kh   = lane >> 4;
  const __bf16* p = tile + (mlocal + (lane & 15)) * APITCH + 8 * kh;
  FragU f;
  f.u[0] = *(const uint4*)(p);        // ds_load_b128
  f.u[1] = *(const uint4*)(p + 16);
  return f.v;
}

__device__ __forceinline__ v16bf lds_frag_b(const __bf16* tile, int nlocal) {
  int lane = threadIdx.x & 31;
  int kh   = lane >> 4;
  const __bf16* p = tile + (nlocal + (lane & 15)) * APITCH + 8 * kh;
  FragU f;
  f.u[0] = *(const uint4*)(p);
  f.u[1] = *(const uint4*)(p + 16);
  return f.v;
}

__device__ __forceinline__ v16bf load_a_f32cvt(const float* __restrict__ A, int lda,
                                               int m0, int k0) {
  int lane = threadIdx.x & 31;
  int kh   = lane >> 4;
  const float* p0 = A + (size_t)(m0 + (lane & 15)) * lda + k0 + 8 * kh;
  v16bf v;
#pragma unroll
  for (int j = 0; j < 8; ++j) v[j] = (__bf16)p0[j];
#pragma unroll
  for (int j = 0; j < 8; ++j) v[j + 8] = (__bf16)p0[16 + j];
  return v;
}

__device__ __forceinline__ float gelu_exact(float x) {
  return 0.5f * x * (1.0f + erff(x * 0.70710678118654752f));
}

// Stage one K-step of tiles into LDS with async copies.
// A tile: 128 rows x 32 bf16 (skipped when A_F32), B tile: 64 rows x 32 bf16.
template <bool A_F32>
__device__ __forceinline__ void stage_tiles(const __bf16* __restrict__ Ab, int lda, int mblk,
                                            const __bf16* __restrict__ W, int ldw, int nblk,
                                            int k0, __bf16* Asm, __bf16* Bsm) {
  int t = threadIdx.x;
  if constexpr (!A_F32) {
#pragma unroll
    for (int i = 0; i < 2; ++i) {
      int c = t * 2 + i;          // 512 chunks of 16B
      int row = c >> 2, q = c & 3;
      async_load_b128(Ab + (size_t)(mblk + row) * lda + k0 + q * 8,
                      Asm + row * APITCH + q * 8);
    }
  }
  {
    int row = t >> 2, q = t & 3;  // 256 chunks of 16B
    async_load_b128(W + (size_t)(nblk + row) * ldw + k0 + q * 8,
                    Bsm + row * APITCH + q * 8);
  }
}

// ---------------------------------------------------------------------------
// Generic batched GEMM:  C[m][n] = act( scale * sum_k A[m][k]*W[n][k] + bias[n] ) (+res)
// grid: (N/64, M/128, Z)   block: 256 threads = 8 waves, wave tile 32x32
// LDS double-buffered tiles filled by GLOBAL_LOAD_ASYNC_TO_LDS_B128.
// ---------------------------------------------------------------------------
template <int ACT, bool A_F32, bool OUT_BF, bool RES>
__global__ __launch_bounds__(256) void gemm_wmma(
    const void* __restrict__ Av, long long sAb, long long sAh, int lda,
    const __bf16* __restrict__ Wv, long long sWb, long long sWh, int ldw,
    void* __restrict__ Cv, long long sCb, long long sCh, int ldc,
    const float* __restrict__ bias, const float* __restrict__ res,
    int K, float scale, int batchH) {
  constexpr int ABUFS = A_F32 ? 1 : 2;
  __shared__ __bf16 Asm[ABUFS][128 * APITCH];
  __shared__ __bf16 Bsm[2][64 * APITCH];

  int z  = blockIdx.z;
  int zb = z / batchH, zh = z % batchH;
  size_t offA = (size_t)zb * sAb + (size_t)zh * sAh;
  size_t offW = (size_t)zb * sWb + (size_t)zh * sWh;
  size_t offC = (size_t)zb * sCb + (size_t)zh * sCh;

  const __bf16* Ab = (const __bf16*)Av + (A_F32 ? 0 : offA);
  const float*  Af = (const float*)Av + (A_F32 ? offA : 0);
  const __bf16* W  = Wv + offW;

  int wave = threadIdx.x >> 5;
  int wm = wave & 3, wn = wave >> 2;
  int mblk = blockIdx.y * 128;
  int nblk = blockIdx.x * 64;
  int m0 = mblk + wm * 32;

  v8f acc[2][2] = {{{0}, {0}}, {{0}, {0}}};

  const int nsteps = K >> 5;
  stage_tiles<A_F32>(Ab, lda, mblk, W, ldw, nblk, 0, &Asm[0][0], &Bsm[0][0]);

  for (int i = 0; i < nsteps; ++i) {
    int k0  = i << 5;
    int cur = i & 1;
    wait_async0();        // this wave's share of the current tile has landed
    __syncthreads();      // whole tile landed; everyone done reading other buffer
    if (i + 1 < nsteps)
      stage_tiles<A_F32>(Ab, lda, mblk, W, ldw, nblk, k0 + 32,
                         &Asm[A_F32 ? 0 : (cur ^ 1)][0], &Bsm[cur ^ 1][0]);
    v16bf a0, a1;
    if constexpr (A_F32) {
      a0 = load_a_f32cvt(Af, lda, m0, k0);
      a1 = load_a_f32cvt(Af, lda, m0 + 16, k0);
    } else {
      a0 = lds_frag_a(&Asm[cur][0], wm * 32);
      a1 = lds_frag_a(&Asm[cur][0], wm * 32 + 16);
    }
    v16bf b0 = lds_frag_b(&Bsm[cur][0], wn * 32);
    v16bf b1 = lds_frag_b(&Bsm[cur][0], wn * 32 + 16);
    acc[0][0] = __builtin_amdgcn_wmma_f32_16x16x32_bf16(false, a0, false, b0, (short)0, acc[0][0], false, false);
    acc[0][1] = __builtin_amdgcn_wmma_f32_16x16x32_bf16(false, a0, false, b1, (short)0, acc[0][1], false, false);
    acc[1][0] = __builtin_amdgcn_wmma_f32_16x16x32_bf16(false, a1, false, b0, (short)0, acc[1][0], false, false);
    acc[1][1] = __builtin_amdgcn_wmma_f32_16x16x32_bf16(false, a1, false, b1, (short)0, acc[1][1], false, false);
  }

  // C/D layout: lane (kh*16+col) holds column n; VGPR v holds row m0 + kh*8 + v
  int lane = threadIdx.x & 31;
  int kh = lane >> 4, col = lane & 15;
  int n0 = nblk + wn * 32;
  float*  Cf = (float*)Cv;
  __bf16* Cb = (__bf16*)Cv;
#pragma unroll
  for (int i = 0; i < 2; ++i) {
#pragma unroll
    for (int j = 0; j < 2; ++j) {
      int n = n0 + j * 16 + col;
      float bb = bias ? bias[n] : 0.0f;
#pragma unroll
      for (int v = 0; v < 8; ++v) {
        int m = m0 + i * 16 + kh * 8 + v;
        float val = acc[i][j][v] * scale + bb;
        if constexpr (ACT == 1) val = gelu_exact(val);
        size_t idx = offC + (size_t)m * ldc + n;
        if constexpr (RES) val += res[idx];
        if constexpr (OUT_BF) Cb[idx] = (__bf16)val;
        else                  Cf[idx] = val;
      }
    }
  }
}

// ---------------------------------------------------------------------------
// LayerNorm (D=1024) -> bf16.  One block (256 thr) per row, 4 elems/thread.
// ---------------------------------------------------------------------------
__device__ __forceinline__ float block_sum(float v, float* red) {
  int t = threadIdx.x;
  red[t] = v; __syncthreads();
  for (int off = 128; off > 0; off >>= 1) {
    if (t < off) red[t] += red[t + off];
    __syncthreads();
  }
  float r = red[0]; __syncthreads();
  return r;
}

__global__ __launch_bounds__(256) void ln_to_bf16(const float* __restrict__ x,
                                                  const float* __restrict__ g,
                                                  const float* __restrict__ be,
                                                  __bf16* __restrict__ out) {
  __shared__ float red[256];
  int row = blockIdx.x;
  const float* xr = x + (size_t)row * DQ;
  float v[4], s = 0.f;
#pragma unroll
  for (int i = 0; i < 4; ++i) { v[i] = xr[threadIdx.x + i * 256]; s += v[i]; }
  float mu = block_sum(s, red) * (1.0f / DQ);
  float q = 0.f;
#pragma unroll
  for (int i = 0; i < 4; ++i) { float d = v[i] - mu; q += d * d; }
  float var = block_sum(q, red) * (1.0f / DQ);
  float rs = rsqrtf(var + 1e-5f);
#pragma unroll
  for (int i = 0; i < 4; ++i) {
    int c = threadIdx.x + i * 256;
    out[(size_t)row * DQ + c] = (__bf16)((v[i] - mu) * rs * g[c] + be[c]);
  }
}

// ---------------------------------------------------------------------------
// Exact top-k(307) + softmax over each 1024-wide score row, in place.
// 32-bit radix select on sign-flipped float bits in LDS.
// ---------------------------------------------------------------------------
__global__ __launch_bounds__(256) void topk_softmax_inplace(float* __restrict__ attn) {
  __shared__ float    sv[SQ];
  __shared__ unsigned su[SQ];
  __shared__ float    fred[256];
  __shared__ int      ired[256];
  float* row = attn + (size_t)blockIdx.x * SQ;
  int t = threadIdx.x;
#pragma unroll
  for (int i = 0; i < 4; ++i) {
    int c = t + i * 256;
    float f = row[c];
    unsigned u = __float_as_uint(f);
    su[c] = (u & 0x80000000u) ? ~u : (u | 0x80000000u);  // monotonic map
    sv[c] = f;
  }
  __syncthreads();
  float m = -3.4e38f;
#pragma unroll
  for (int i = 0; i < 4; ++i) m = fmaxf(m, sv[t + i * 256]);
  fred[t] = m; __syncthreads();
  for (int off = 128; off > 0; off >>= 1) {
    if (t < off) fred[t] = fmaxf(fred[t], fred[t + off]);
    __syncthreads();
  }
  float rowmax = fred[0]; __syncthreads();
  unsigned prefix = 0u;
  for (int bit = 31; bit >= 0; --bit) {
    unsigned trial = prefix | (1u << bit);
    int c = 0;
#pragma unroll
    for (int i = 0; i < 4; ++i) c += (su[t + i * 256] >= trial) ? 1 : 0;
    ired[t] = c; __syncthreads();
    for (int off = 128; off > 0; off >>= 1) {
      if (t < off) ired[t] += ired[t + off];
      __syncthreads();
    }
    if (ired[0] >= TOPK) prefix = trial;
    __syncthreads();
  }
  float s = 0.f;
#pragma unroll
  for (int i = 0; i < 4; ++i) {
    int c = t + i * 256;
    if (su[c] >= prefix) s += __expf(sv[c] - rowmax);
  }
  fred[t] = s; __syncthreads();
  for (int off = 128; off > 0; off >>= 1) {
    if (t < off) fred[t] += fred[t + off];
    __syncthreads();
  }
  float inv = 1.0f / fred[0]; __syncthreads();
#pragma unroll
  for (int i = 0; i < 4; ++i) {
    int c = t + i * 256;
    row[c] = (su[c] >= prefix) ? __expf(sv[c] - rowmax) * inv : 0.0f;
  }
}

// ---------------------------------------------------------------------------
// Helpers: f32 -> bf16 convert, and V -> Vt per (b,h):  Vt[bh][dk][s] = V[b,s,h*64+dk]
// ---------------------------------------------------------------------------
__global__ __launch_bounds__(256) void f32_to_bf16(const float* __restrict__ s,
                                                   __bf16* __restrict__ d, int n4) {
  int i = blockIdx.x * 256 + threadIdx.x;
  if (i < n4) {
    float4 f = ((const float4*)s)[i];
    d[i * 4 + 0] = (__bf16)f.x;
    d[i * 4 + 1] = (__bf16)f.y;
    d[i * 4 + 2] = (__bf16)f.z;
    d[i * 4 + 3] = (__bf16)f.w;
  }
}

__global__ __launch_bounds__(256) void build_vt(const __bf16* __restrict__ V,
                                                __bf16* __restrict__ Vt) {
  int idx = blockIdx.x * 256 + threadIdx.x;  // over BQ*HQ*DKQ*SQ
  int s  = idx & (SQ - 1);
  int r  = idx >> 10;
  int dk = r & (DKQ - 1);
  int bh = r >> 6;
  int h  = bh & (HQ - 1);
  int b  = bh >> 4;
  Vt[idx] = V[((size_t)(b * SQ + s)) * DQ + h * DKQ + dk];
}

// ---------------------------------------------------------------------------
extern "C" void kernel_launch(void* const* d_in, const int* in_sizes, int n_in,
                              void* d_out, int out_size, void* d_ws, size_t ws_size,
                              hipStream_t stream) {
  (void)in_sizes; (void)n_in; (void)out_size; (void)ws_size;
  const float* x  = (const float*)d_in[0];
  const float* Wq = (const float*)d_in[1];  const float* bq = (const float*)d_in[2];
  const float* Wk = (const float*)d_in[3];  const float* bk = (const float*)d_in[4];
  const float* Wv = (const float*)d_in[5];  const float* bv = (const float*)d_in[6];
  const float* Wo = (const float*)d_in[7];  const float* bo = (const float*)d_in[8];
  const float* W1 = (const float*)d_in[9];  const float* b1 = (const float*)d_in[10];
  const float* Wm = (const float*)d_in[11]; const float* bm = (const float*)d_in[12];
  const float* W2 = (const float*)d_in[13]; const float* b2 = (const float*)d_in[14];
  const float* g1 = (const float*)d_in[15]; const float* be1 = (const float*)d_in[16];
  const float* g2 = (const float*)d_in[17]; const float* be2 = (const float*)d_in[18];

  float* out_x = (float*)d_out;
  float* attn  = out_x + (size_t)BQ * SQ * DQ;  // second tuple output

  const int MT = BQ * SQ;  // 4096 token rows

  char* p = (char*)d_ws;
  auto carve = [&](size_t bytes) {
    char* r = p;
    p += (bytes + 255) & ~(size_t)255;
    return r;
  };
  __bf16* wq_bf = (__bf16*)carve((size_t)DQ * DQ * 2);
  __bf16* wk_bf = (__bf16*)carve((size_t)DQ * DQ * 2);
  __bf16* wv_bf = (__bf16*)carve((size_t)DQ * DQ * 2);
  __bf16* wo_bf = (__bf16*)carve((size_t)DQ * DQ * 2);
  __bf16* w1_bf = (__bf16*)carve((size_t)DFQ * DQ * 2);
  __bf16* wm_bf = (__bf16*)carve((size_t)DFQ * DFQ * 2);
  __bf16* w2_bf = (__bf16*)carve((size_t)DQ * DFQ * 2);
  __bf16* h1    = (__bf16*)carve((size_t)MT * DQ * 2);
  __bf16* Qb    = (__bf16*)carve((size_t)MT * DQ * 2);
  __bf16* Kb    = (__bf16*)carve((size_t)MT * DQ * 2);
  __bf16* Vb    = (__bf16*)carve((size_t)MT * DQ * 2);
  __bf16* Vt    = (__bf16*)carve((size_t)MT * DQ * 2);
  __bf16* ctx   = (__bf16*)carve((size_t)MT * DQ * 2);
  float*  x1    = (float*)carve((size_t)MT * DQ * 4);
  __bf16* h2    = (__bf16*)carve((size_t)MT * DQ * 2);
  __bf16* f1    = (__bf16*)carve((size_t)MT * DFQ * 2);
  __bf16* f2    = (__bf16*)carve((size_t)MT * DFQ * 2);

  auto cvt = [&](const float* s, __bf16* d, size_t n) {
    int n4 = (int)(n / 4);
    f32_to_bf16<<<(n4 + 255) / 256, 256, 0, stream>>>(s, d, n4);
  };
  cvt(Wq, wq_bf, (size_t)DQ * DQ);
  cvt(Wk, wk_bf, (size_t)DQ * DQ);
  cvt(Wv, wv_bf, (size_t)DQ * DQ);
  cvt(Wo, wo_bf, (size_t)DQ * DQ);
  cvt(W1, w1_bf, (size_t)DFQ * DQ);
  cvt(Wm, wm_bf, (size_t)DFQ * DFQ);
  cvt(W2, w2_bf, (size_t)DQ * DFQ);

  ln_to_bf16<<<MT, 256, 0, stream>>>(x, g1, be1, h1);

  dim3 gQKV(DQ / 64, MT / 128, 1);
  gemm_wmma<0, false, true, false><<<gQKV, 256, 0, stream>>>(
      h1, 0, 0, DQ, wq_bf, 0, 0, DQ, Qb, 0, 0, DQ, bq, nullptr, DQ, 1.0f, 1);
  gemm_wmma<0, false, true, false><<<gQKV, 256, 0, stream>>>(
      h1, 0, 0, DQ, wk_bf, 0, 0, DQ, Kb, 0, 0, DQ, bk, nullptr, DQ, 1.0f, 1);
  gemm_wmma<0, false, true, false><<<gQKV, 256, 0, stream>>>(
      h1, 0, 0, DQ, wv_bf, 0, 0, DQ, Vb, 0, 0, DQ, bv, nullptr, DQ, 1.0f, 1);

  {
    int total = BQ * HQ * DKQ * SQ;
    build_vt<<<total / 256, 256, 0, stream>>>(Vb, Vt);
  }

  // scores = Q K^T / 8, written straight into the attn output region
  {
    dim3 g(SQ / 64, SQ / 128, BQ * HQ);
    gemm_wmma<0, false, false, false><<<g, 256, 0, stream>>>(
        Qb, (long long)SQ * DQ, DKQ, DQ,
        Kb, (long long)SQ * DQ, DKQ, DQ,
        attn, (long long)HQ * SQ * SQ, (long long)SQ * SQ, SQ,
        nullptr, nullptr, DKQ, 0.125f, HQ);
  }

  topk_softmax_inplace<<<BQ * HQ * SQ, 256, 0, stream>>>(attn);

  // ctx = attn @ V  (A fp32 converted per fragment; B = Vt staged via async LDS)
  {
    dim3 g(DKQ / 64, SQ / 128, BQ * HQ);
    gemm_wmma<0, true, true, false><<<g, 256, 0, stream>>>(
        attn, (long long)HQ * SQ * SQ, (long long)SQ * SQ, SQ,
        Vt, (long long)HQ * DKQ * SQ, (long long)DKQ * SQ, SQ,
        ctx, (long long)SQ * DQ, DKQ, DQ,
        nullptr, nullptr, SQ, 1.0f, HQ);
  }

  // x1 = x + ctx @ Wo^T + bo
  gemm_wmma<0, false, false, true><<<gQKV, 256, 0, stream>>>(
      ctx, 0, 0, DQ, wo_bf, 0, 0, DQ, x1, 0, 0, DQ, bo, x, DQ, 1.0f, 1);

  ln_to_bf16<<<MT, 256, 0, stream>>>(x1, g2, be2, h2);

  // f1 = gelu(h2 @ W1^T + b1)
  {
    dim3 g(DFQ / 64, MT / 128, 1);
    gemm_wmma<1, false, true, false><<<g, 256, 0, stream>>>(
        h2, 0, 0, DQ, w1_bf, 0, 0, DQ, f1, 0, 0, DFQ, b1, nullptr, DQ, 1.0f, 1);
  }
  // f2 = gelu(f1 @ Wm^T + bm)  — dominant GEMM
  {
    dim3 g(DFQ / 64, MT / 128, 1);
    gemm_wmma<1, false, true, false><<<g, 256, 0, stream>>>(
        f1, 0, 0, DFQ, wm_bf, 0, 0, DFQ, f2, 0, 0, DFQ, bm, nullptr, DFQ, 1.0f, 1);
  }
  // out_x = x1 + f2 @ W2^T + b2
  {
    dim3 g(DQ / 64, MT / 128, 1);
    gemm_wmma<0, false, false, true><<<g, 256, 0, stream>>>(
        f2, 0, 0, DFQ, w2_bf, 0, 0, DFQ, out_x, 0, 0, DQ, b2, x1, DFQ, 1.0f, 1);
  }
}
